// NTXentLoss_modify_16157666967722
// MI455X (gfx1250) — matrix operand order repaired and checked by reading, not compiled, not run on hardware
//
#include <hip/hip_runtime.h>
#include <hip/hip_bf16.h>

// NT-Xent loss, fused: normalize -> f16 rep -> tiled WMMA sim + streaming
// exp-sum (never materializing the 8192x8192 sim matrix) -> per-row loss.
// B=4096, D=256, T=0.1. Deterministic (no atomics).
// CDNA5 paths: v_wmma_f32_16x16x32_f16, global_load_async_to_lds_b128,
// s_wait_asynccnt-driven LDS double buffering.

#define BB      4096
#define NN      8192            // 2B rows of rep
#define DD      256             // embedding dim
#define CSPLIT  8               // column chunks per row block (parallelism)
#define ST      32              // columns per LDS super-tile
#define NST     (NN / ST)       // 256 super-tiles
#define NT_PER  (NST / CSPLIT)  // 32 super-tiles per chunk
#define LSTRIDE 264             // LDS row stride in halves (256 + 8 pad -> conflict-free)

typedef __attribute__((ext_vector_type(16))) _Float16 v16h;
typedef __attribute__((ext_vector_type(8)))  _Float16 v8h;
typedef __attribute__((ext_vector_type(8)))  float    v8f;
typedef __attribute__((ext_vector_type(4)))  float    v4f;

union V16 { v16h v; v8h h[2]; };

// Async DMA: 16 bytes global -> LDS, tracked by ASYNCcnt (no staging VGPRs).
__device__ __forceinline__ void async_ld_b128(unsigned lds_off, const _Float16* g) {
  asm volatile("global_load_async_to_lds_b128 %0, %1, off"
               :: "v"(lds_off), "v"(g) : "memory");
}
#define S_WAIT_ASYNC(N) asm volatile("s_wait_asynccnt " #N ::: "memory")

// ---------------------------------------------------------------------------
// Kernel 1: L2-normalize rows of [zjs; zis] and store as f16 rep (NN x DD).
// ---------------------------------------------------------------------------
__global__ __launch_bounds__(256) void k_normalize(const float* __restrict__ zis,
                                                   const float* __restrict__ zjs,
                                                   _Float16* __restrict__ rep) {
  const int row  = (blockIdx.x * 256 + threadIdx.x) >> 5;
  const int lane = threadIdx.x & 31;
  const float* src = (row < BB) ? (zjs + (size_t)row * DD)
                                : (zis + (size_t)(row - BB) * DD);
  v4f x0 = *(const v4f*)(src + lane * 8);
  v4f x1 = *(const v4f*)(src + lane * 8 + 4);
  float ss = 0.f;
#pragma unroll
  for (int j = 0; j < 4; ++j) ss += x0[j] * x0[j] + x1[j] * x1[j];
#pragma unroll
  for (int off = 1; off < 32; off <<= 1) ss += __shfl_xor(ss, off, 32);
  const float scale = 1.0f / fmaxf(sqrtf(ss), 1e-12f);
  v8h o;
#pragma unroll
  for (int j = 0; j < 4; ++j) {
    o[j]     = (_Float16)(x0[j] * scale);
    o[4 + j] = (_Float16)(x1[j] * scale);
  }
  *(v8h*)(rep + (size_t)row * DD + lane * 8) = o;
}

// ---------------------------------------------------------------------------
// Kernel 2: fused sim GEMM + exp row sums.
// Block = 8 waves; wave owns rows r0..r0+15, A fragments resident in VGPRs
// (ISA 16x32 f16 A layout). Block shares 32-column B super-tiles staged by
// async-to-LDS DMA, double buffered on ASYNCcnt. Each wave runs TWO
// independent WMMA accumulator chains (cols +0..15 and +16..31) for ILP.
// ---------------------------------------------------------------------------
__global__ __launch_bounds__(256) void k_simexp(const _Float16* __restrict__ rep,
                                                float* __restrict__ chunkS) {
  const int tid  = threadIdx.x;
  const int lane = tid & 31;
  const int wave = tid >> 5;
  const int l    = lane & 15;   // A row / B column within 16x16 tile
  const int grp  = lane >> 4;   // half-wave group
  const int r0   = blockIdx.x * 128 + wave * 16;
  const int st0  = blockIdx.y * NT_PER;

  __shared__ _Float16 lds[2][ST * LSTRIDE];   // 2 x 16.5 KB

  // Resident A fragments: lane l<16 holds row r0+l, halves {K..K+7, K+16..K+23}.
  v16h a[8];
  const _Float16* arow = rep + (size_t)(r0 + l) * DD;
#pragma unroll
  for (int kk = 0; kk < 8; ++kk) {
    const int base = kk * 32 + grp * 8;
    V16 u;
    u.h[0] = *(const v8h*)(arow + base);
    u.h[1] = *(const v8h*)(arow + base + 16);
    a[kk] = u.v;
  }

  float acc[8];
#pragma unroll
  for (int v = 0; v < 8; ++v) acc[v] = 0.f;

  // Staging roles: thread -> (tile row 0..31, 64-byte chunk 0..7).
  const int trow = tid >> 3;
  const int tch  = (tid & 7) * 32;            // halves
  const unsigned lds0      = (unsigned)(uintptr_t)(&lds[0][0]);
  const unsigned myoff     = (unsigned)((trow * LSTRIDE + tch) * 2);
  const unsigned bufstride = (unsigned)(ST * LSTRIDE * 2);

  auto issue = [&](int t) {
    const _Float16* g = rep + (size_t)((st0 + t) * ST + trow) * DD + tch;
    const unsigned dst = lds0 + (unsigned)(t & 1) * bufstride + myoff;
    async_ld_b128(dst +  0, g +  0);
    async_ld_b128(dst + 16, g +  8);
    async_ld_b128(dst + 32, g + 16);
    async_ld_b128(dst + 48, g + 24);
  };

  issue(0);
  for (int t = 0; t < NT_PER; ++t) {
    if (t + 1 < NT_PER) { issue(t + 1); S_WAIT_ASYNC(4); }  // tile t arrived (mine)
    else                { S_WAIT_ASYNC(0); }
    __syncthreads();    // tile t arrived (all waves)

    const _Float16* base = &lds[t & 1][0];
    const _Float16* bb0 = base + (l     ) * LSTRIDE + grp * 8;
    const _Float16* bb1 = base + (l + 16) * LSTRIDE + grp * 8;
    v8f c0 = {0.f, 0.f, 0.f, 0.f, 0.f, 0.f, 0.f, 0.f};
    v8f c1 = {0.f, 0.f, 0.f, 0.f, 0.f, 0.f, 0.f, 0.f};
#pragma unroll
    for (int kk = 0; kk < 8; ++kk) {
      V16 u0, u1;
      u0.h[0] = *(const v8h*)(bb0 + kk * 32);
      u0.h[1] = *(const v8h*)(bb0 + kk * 32 + 16);
      u1.h[0] = *(const v8h*)(bb1 + kk * 32);
      u1.h[1] = *(const v8h*)(bb1 + kk * 32 + 16);
      c0 = __builtin_amdgcn_wmma_f32_16x16x32_f16(false, a[kk], false, u0.v,
                                                  (short)0, c0, false, false);
      c1 = __builtin_amdgcn_wmma_f32_16x16x32_f16(false, a[kk], false, u1.v,
                                                  (short)0, c1, false, false);
    }
    // C/D layout: VGPR v -> row (r0 + grp*8 + v).
#pragma unroll
    for (int v = 0; v < 8; ++v)
      acc[v] += __expf(c0[v] * 10.0f - 10.0f) + __expf(c1[v] * 10.0f - 10.0f);

    __syncthreads();    // all waves done reading buf before it is re-filled
  }

  // Reduce across the 16 lanes of each half-wave (offsets < 16 stay in group).
#pragma unroll
  for (int v = 0; v < 8; ++v) {
#pragma unroll
    for (int off = 1; off < 16; off <<= 1) acc[v] += __shfl_xor(acc[v], off, 32);
  }
  if (l == 0) {
#pragma unroll
    for (int v = 0; v < 8; ++v)
      chunkS[(size_t)blockIdx.y * NN + (r0 + grp * 8 + v)] = acc[v];
  }
}

// ---------------------------------------------------------------------------
// Kernel 3: per-row loss. pos/diag dot products from f16 rep (one wave/row),
// combine chunk sums: loss = -10*pos + log(S - exp(10*diag - 10)) + 10.
// ---------------------------------------------------------------------------
__global__ __launch_bounds__(256) void k_rowloss(const _Float16* __restrict__ rep,
                                                 const float* __restrict__ chunkS,
                                                 float* __restrict__ rowLoss) {
  const int row  = (blockIdx.x * 256 + threadIdx.x) >> 5;
  const int lane = threadIdx.x & 31;
  const int partner = row ^ BB;  // (row + B) mod 2B
  v8h av = ((const v8h*)(rep + (size_t)row * DD))[lane];
  v8h bv = ((const v8h*)(rep + (size_t)partner * DD))[lane];
  float pos = 0.f, dg = 0.f;
#pragma unroll
  for (int j = 0; j < 8; ++j) {
    const float x = (float)av[j], y = (float)bv[j];
    pos += x * y;
    dg  += x * x;
  }
#pragma unroll
  for (int off = 1; off < 32; off <<= 1) {
    pos += __shfl_xor(pos, off, 32);
    dg  += __shfl_xor(dg, off, 32);
  }
  if (lane == 0) {
    float S = 0.f;
#pragma unroll
    for (int c = 0; c < CSPLIT; ++c) S += chunkS[(size_t)c * NN + row];
    rowLoss[row] = -10.0f * pos + logf(S - __expf(10.0f * dg - 10.0f)) + 10.0f;
  }
}

// ---------------------------------------------------------------------------
// Kernel 4: deterministic reduction of NN row losses -> out[0] = sum / NN.
// ---------------------------------------------------------------------------
__global__ __launch_bounds__(256) void k_reduce(const float* __restrict__ rowLoss,
                                                float* __restrict__ out) {
  __shared__ float sbuf[8];
  const int tid = threadIdx.x, lane = tid & 31, wave = tid >> 5;
  float s = 0.f;
  for (int i = tid; i < NN; i += 256) s += rowLoss[i];
#pragma unroll
  for (int off = 1; off < 32; off <<= 1) s += __shfl_xor(s, off, 32);
  if (lane == 0) sbuf[wave] = s;
  __syncthreads();
  if (tid == 0) {
    float tot = 0.f;
#pragma unroll
    for (int w = 0; w < 8; ++w) tot += sbuf[w];
    out[0] = tot / (float)NN;
  }
}

// ---------------------------------------------------------------------------
extern "C" void kernel_launch(void* const* d_in, const int* in_sizes, int n_in,
                              void* d_out, int out_size, void* d_ws, size_t ws_size,
                              hipStream_t stream) {
  (void)in_sizes; (void)n_in; (void)out_size; (void)ws_size;
  const float* zis = (const float*)d_in[0];  // rep rows B..2B-1
  const float* zjs = (const float*)d_in[1];  // rep rows 0..B-1
  // d_in[2], d_in[3] (x1, x2) unused by the reference forward.

  _Float16* rep     = (_Float16*)d_ws;                              // 4 MB
  float*    chunkS  = (float*)((char*)d_ws + (size_t)NN * DD * 2);  // 256 KB
  float*    rowLoss = chunkS + (size_t)CSPLIT * NN;                 // 32 KB
  float*    out     = (float*)d_out;

  k_normalize<<<NN / 8, 256, 0, stream>>>(zis, zjs, rep);
  dim3 g2(NN / 128, CSPLIT);
  k_simexp<<<g2, 256, 0, stream>>>(rep, chunkS);
  k_rowloss<<<NN / 8, 256, 0, stream>>>(rep, chunkS, rowLoss);
  k_reduce<<<1, 256, 0, stream>>>(rowLoss, out);
}
